// CA_41936060678207
// MI455X (gfx1250) — matrix operand order repaired and checked by reading, not compiled, not run on hardware
//
#include <hip/hip_runtime.h>
#include <hip/hip_bf16.h>
#include <stdint.h>

typedef __attribute__((ext_vector_type(16))) __bf16 v16bf;
typedef __attribute__((ext_vector_type(8)))  float  v8f;
typedef __attribute__((ext_vector_type(4)))  int    v4i;

#define D_IN  256
#define D_HID 512
#define D_OUT 256
#define BB    4
#define MMR   4096
#define ROWS  (BB * MMR)      // 16384
#define TEMPF 10.0f

// ---------------------------------------------------------------------------
// CDNA5 async Global->LDS path (ASYNCcnt), with safe fallbacks.
// param0: v4i* in AS(1) (global, non-const); param1: LDS-side ptr (AS(3));
// then two int immediates (offset, cpol).
// ---------------------------------------------------------------------------
#if defined(__has_builtin)
#  if __has_builtin(__builtin_amdgcn_global_load_async_to_lds_b128)
#    define HAVE_ASYNC_LD 1
#  else
#    define HAVE_ASYNC_LD 0
#  endif
#  if __has_builtin(__builtin_amdgcn_global_store_async_from_lds_b128)
#    define HAVE_ASYNC_ST 1
#  else
#    define HAVE_ASYNC_ST 0
#  endif
#  if __has_builtin(__builtin_amdgcn_s_wait_asynccnt)
#    define HAVE_WAIT_ASYNC 1
#  else
#    define HAVE_WAIT_ASYNC 0
#  endif
#else
#  define HAVE_ASYNC_LD 0
#  define HAVE_ASYNC_ST 0
#  define HAVE_WAIT_ASYNC 0
#endif

typedef __attribute__((address_space(1))) v4i* gv4p;   // global int4*
typedef __attribute__((address_space(3))) v4i* lv4p;   // LDS int4*

__device__ __forceinline__ void async_copy16(const void* gsrc, void* ldst) {
#if HAVE_ASYNC_LD
    __builtin_amdgcn_global_load_async_to_lds_b128((gv4p)gsrc, (lv4p)ldst, 0, 0);
#else
    *(uint4*)ldst = *(const uint4*)gsrc;
#endif
}

__device__ __forceinline__ void async_store16(void* gdst, const void* lsrc) {
#if HAVE_ASYNC_ST
    __builtin_amdgcn_global_store_async_from_lds_b128((gv4p)gdst, (lv4p)lsrc, 0, 0);
#else
    *(uint4*)gdst = *(const uint4*)lsrc;
#endif
}

__device__ __forceinline__ void wait_async0() {
#if HAVE_ASYNC_LD || HAVE_ASYNC_ST
#  if HAVE_WAIT_ASYNC
    __builtin_amdgcn_s_wait_asynccnt(0);
#  else
    asm volatile("s_wait_asynccnt 0x0" ::: "memory");
#  endif
#endif
}

// ---------------------------------------------------------------------------
// f32 -> bf16 elementwise convert
// ---------------------------------------------------------------------------
__global__ void cvt_bf16_kernel(const float* __restrict__ in,
                                __bf16* __restrict__ out, int n) {
    int i = blockIdx.x * blockDim.x + threadIdx.x;
    if (i < n) out[i] = (__bf16)in[i];
}

// ---------------------------------------------------------------------------
// f32 [K,N] -> bf16 [N,K] transpose convert (weights are tiny, L2-resident)
// ---------------------------------------------------------------------------
__global__ void cvt_t_bf16_kernel(const float* __restrict__ in,
                                  __bf16* __restrict__ out, int K, int N) {
    int i = blockIdx.x * blockDim.x + threadIdx.x;
    if (i < K * N) {
        int k = i / N, n = i - k * N;
        out[(size_t)n * K + k] = (__bf16)in[i];
    }
}

// ---------------------------------------------------------------------------
// bf16 WMMA GEMM with LDS-staged B strip and software-pipelined inner loop:
//   C[M,N] = act(A[M,K] @ BT[N,K]^T + bias)
// Block = 256 threads = 8 waves, block tile = 128 rows x 64 cols.
// The 64 x K bf16 strip of BT is async-copied to LDS once (ASYNCcnt) and
// shared by all 8 waves. Per K-step: all 4 B fragments are ds_load'ed into
// DISTINCT registers and both A (global) and B (LDS) fragments are
// double-buffered across K-steps, so one wait covers a clause of loads and
// the 4 v_wmma_f32_16x16x32_bf16 issue back-to-back while the next step's
// loads are in flight. blockIdx.y = batch (strides in elements).
// ---------------------------------------------------------------------------
template <int K, bool RELU, bool STORE_BF16>
__global__ __launch_bounds__(256)
void wmma_gemm_kernel(const __bf16* __restrict__ A,
                      const __bf16* __restrict__ BT,
                      const float* __restrict__ bias,
                      void* __restrict__ Cout,
                      int M, int N,
                      size_t strideA, size_t strideB, size_t strideC) {
    constexpr int NB   = 4;          // 4 N-tiles of 16 per wave
    constexpr int BN   = 64;         // block N extent
    constexpr int KP   = K + 8;      // LDS row pitch (pad to dodge bank camping)
    __shared__ __align__(16) __bf16 Bs[BN][KP];

    const int lane = threadIdx.x & 31;
    const int wave = threadIdx.x >> 5;

    A  += (size_t)blockIdx.y * strideA;
    BT += (size_t)blockIdx.y * strideB;

    const int nBlocks = N / BN;
    const int bm = blockIdx.x / nBlocks;     // 128-row block index
    const int bn = blockIdx.x % nBlocks;     // 64-col block index
    const int n0 = bn * BN;

    // ---- stage BT[n0 .. n0+63][0..K) into LDS via async copies ----
    {
        constexpr int CH = K / 8;            // 16B chunks per row
        const __bf16* bsrc = BT + (size_t)n0 * K;
        for (int i = threadIdx.x; i < BN * CH; i += 256) {
            const int row = i / CH;
            const int c   = i - row * CH;
            async_copy16(bsrc + (size_t)row * K + c * 8, &Bs[row][c * 8]);
        }
        wait_async0();
        __syncthreads();
    }

    const int tm = bm * 8 + wave;            // this wave's 16-row tile
    const int r  = lane & 15;
    const int hi = lane >> 4;

    v8f zero = {};
    v8f acc[NB];
#pragma unroll
    for (int j = 0; j < NB; ++j) acc[j] = zero;

    const __bf16* arow = A + (size_t)(tm * 16 + r) * K;

    union U { v16bf v; uint4 u[2]; };

    // fragment loaders (k0 is always a compile-time constant after unroll)
    auto loadA = [&](U& a, int k0) {
        a.u[0] = *(const uint4*)(arow + k0 + hi * 8);
        a.u[1] = *(const uint4*)(arow + k0 + 16 + hi * 8);
    };
    auto loadB = [&](U (&b)[NB], int k0) {
#pragma unroll
        for (int j = 0; j < NB; ++j) {
            b[j].u[0] = *(const uint4*)&Bs[j * 16 + r][k0 + hi * 8];
            b[j].u[1] = *(const uint4*)&Bs[j * 16 + r][k0 + 16 + hi * 8];
        }
    };

    // ---- main loop: double-buffered A (global) and B (LDS) fragments ----
    U a_buf[2];
    U b_buf[2][NB];
    loadA(a_buf[0], 0);
    loadB(b_buf[0], 0);
#pragma unroll
    for (int k0 = 0; k0 < K; k0 += 32) {
        const int cur = (k0 >> 5) & 1;
        const int nxt = cur ^ 1;
        if (k0 + 32 < K) {
            loadA(a_buf[nxt], k0 + 32);
            loadB(b_buf[nxt], k0 + 32);
        }
#pragma unroll
        for (int j = 0; j < NB; ++j) {
            acc[j] = __builtin_amdgcn_wmma_f32_16x16x32_bf16(
                false, a_buf[cur].v, false, b_buf[cur][j].v,
                (short)0, acc[j], false, false);
        }
    }

    // ---- epilogue: C/D layout -> VGPR v holds M = v + 8*hi, N = lane&15 ----
#pragma unroll
    for (int j = 0; j < NB; ++j) {
        const int n = n0 + j * 16 + (lane & 15);
        const float bv = bias ? bias[n] : 0.0f;
#pragma unroll
        for (int v = 0; v < 8; ++v) {
            const int m = tm * 16 + v + (hi ? 8 : 0);
            float val = acc[j][v] + bv;
            if (RELU) val = val > 0.0f ? val : 0.0f;
            const size_t cidx = (size_t)blockIdx.y * strideC + (size_t)m * N + n;
            if (STORE_BF16) ((__bf16*)Cout)[cidx] = (__bf16)val;
            else            ((float*)Cout)[cidx]  = val;
        }
    }
}

// ---------------------------------------------------------------------------
// Row softmax with row-max scaling (reference semantics):
//   m = max(row); s = TEMP * row / m; out = softmax(s)
// One 256-thread block per row of 4096 f32. Row staged to LDS via the async
// path (16 KB), all passes run on LDS, result streamed back from LDS.
// ---------------------------------------------------------------------------
__global__ __launch_bounds__(256)
void softmax_kernel(float* __restrict__ out) {
    __shared__ __align__(16) float rowbuf[MMR];
    __shared__ float red[256];

    float* p = out + (size_t)blockIdx.x * MMR;
    const int t = threadIdx.x;

    // stage row into LDS (1024 x 16B chunks, 4 per thread)
    for (int c = t; c < MMR / 4; c += 256)
        async_copy16(p + c * 4, &rowbuf[c * 4]);
    wait_async0();
    __syncthreads();

    // pass 1: row max
    float lmax = -INFINITY;
    for (int i = t; i < MMR; i += 256) lmax = fmaxf(lmax, rowbuf[i]);
    red[t] = lmax;
    __syncthreads();
    for (int s = 128; s > 0; s >>= 1) {
        if (t < s) red[t] = fmaxf(red[t], red[t + s]);
        __syncthreads();
    }
    const float scale = TEMPF / red[0];
    __syncthreads();

    // pass 2: scale + max of scaled (handles negative row-max flipping order)
    float lsmax = -INFINITY;
    for (int i = t; i < MMR; i += 256) {
        float s = rowbuf[i] * scale;
        rowbuf[i] = s;
        lsmax = fmaxf(lsmax, s);
    }
    red[t] = lsmax;
    __syncthreads();
    for (int s = 128; s > 0; s >>= 1) {
        if (t < s) red[t] = fmaxf(red[t], red[t + s]);
        __syncthreads();
    }
    const float smax = red[0];
    __syncthreads();

    // pass 3: exp + sum
    float lsum = 0.0f;
    for (int i = t; i < MMR; i += 256) {
        float e = __expf(rowbuf[i] - smax);
        rowbuf[i] = e;
        lsum += e;
    }
    red[t] = lsum;
    __syncthreads();
    for (int s = 128; s > 0; s >>= 1) {
        if (t < s) red[t] = red[t] + red[t + s];
        __syncthreads();
    }
    const float inv = 1.0f / red[0];
    __syncthreads();

    // normalize in LDS, then stream back via async store-from-LDS
    for (int i = t; i < MMR; i += 256) rowbuf[i] *= inv;
    __syncthreads();
    for (int c = t; c < MMR / 4; c += 256)
        async_store16(p + c * 4, &rowbuf[c * 4]);
    wait_async0();
}

// ---------------------------------------------------------------------------
// Host launcher
// ---------------------------------------------------------------------------
extern "C" void kernel_launch(void* const* d_in, const int* in_sizes, int n_in,
                              void* d_out, int out_size, void* d_ws, size_t ws_size,
                              hipStream_t stream) {
    const float* x   = (const float*)d_in[0];
    const float* Wq1 = (const float*)d_in[1];
    const float* bq1 = (const float*)d_in[2];
    const float* Wq2 = (const float*)d_in[3];
    const float* bq2 = (const float*)d_in[4];
    const float* Wk1 = (const float*)d_in[5];
    const float* bk1 = (const float*)d_in[6];
    const float* Wk2 = (const float*)d_in[7];
    const float* bk2 = (const float*)d_in[8];
    float* out = (float*)d_out;

    // Carve workspace (256B aligned): everything here is L2-resident (~43 MB).
    char* ws = (char*)d_ws;
    size_t off = 0;
    auto carve = [&](size_t bytes) -> char* {
        char* p = ws + off;
        off = (off + bytes + 255) & ~(size_t)255;
        return p;
    };
    __bf16* xb   = (__bf16*)carve((size_t)ROWS * D_IN * 2);
    __bf16* W1qT = (__bf16*)carve((size_t)D_HID * D_IN * 2);
    __bf16* W2qT = (__bf16*)carve((size_t)D_OUT * D_HID * 2);
    __bf16* W1kT = (__bf16*)carve((size_t)D_HID * D_IN * 2);
    __bf16* W2kT = (__bf16*)carve((size_t)D_OUT * D_HID * 2);
    __bf16* H    = (__bf16*)carve((size_t)ROWS * D_HID * 2);
    __bf16* Qb   = (__bf16*)carve((size_t)ROWS * D_OUT * 2);
    __bf16* Kb   = (__bf16*)carve((size_t)ROWS * D_OUT * 2);

    // Stage 0: converts / transposes
    cvt_bf16_kernel<<<(ROWS * D_IN) / 256, 256, 0, stream>>>(x, xb, ROWS * D_IN);
    cvt_t_bf16_kernel<<<(D_IN * D_HID + 255) / 256, 256, 0, stream>>>(Wq1, W1qT, D_IN, D_HID);
    cvt_t_bf16_kernel<<<(D_HID * D_OUT + 255) / 256, 256, 0, stream>>>(Wq2, W2qT, D_HID, D_OUT);
    cvt_t_bf16_kernel<<<(D_IN * D_HID + 255) / 256, 256, 0, stream>>>(Wk1, W1kT, D_IN, D_HID);
    cvt_t_bf16_kernel<<<(D_HID * D_OUT + 255) / 256, 256, 0, stream>>>(Wk2, W2kT, D_HID, D_OUT);

    // Block tile = 128 rows x 64 cols, 8 waves/block.
    // GEMM1: [16384,256] @ [256,512] -> H : (16384/128)*(512/64) = 1024 blocks
    wmma_gemm_kernel<D_IN, true, true><<<dim3(1024, 1), 256, 0, stream>>>(
        xb, W1qT, bq1, H, ROWS, D_HID, 0, 0, 0);
    // GEMM2: [16384,512] @ [512,256] -> Qb : 128*4 = 512 blocks
    wmma_gemm_kernel<D_HID, false, true><<<dim3(512, 1), 256, 0, stream>>>(
        H, W2qT, bq2, Qb, ROWS, D_OUT, 0, 0, 0);
    // K branch (reuses H; stream-ordered after Q branch read it)
    wmma_gemm_kernel<D_IN, true, true><<<dim3(1024, 1), 256, 0, stream>>>(
        xb, W1kT, bk1, H, ROWS, D_HID, 0, 0, 0);
    wmma_gemm_kernel<D_HID, false, true><<<dim3(512, 1), 256, 0, stream>>>(
        H, W2kT, bk2, Kb, ROWS, D_OUT, 0, 0, 0);

    // QK^T per batch: [4096,256] @ [4096,256]^T -> out f32 [4096,4096]
    // per-batch blocks = (4096/128)*(4096/64) = 32*64 = 2048; grid.y = batch
    wmma_gemm_kernel<D_OUT, false, false><<<dim3(2048, BB), 256, 0, stream>>>(
        Qb, Kb, nullptr, out, MMR, MMR,
        (size_t)MMR * D_OUT, (size_t)MMR * D_OUT, (size_t)MMR * MMR);

    // Softmax with row-max scaling, in place on d_out
    softmax_kernel<<<ROWS, 256, 0, stream>>>(out);
}
